// CausalSelfAttention_66666482369242
// MI455X (gfx1250) — compile-verified, matrix operand order checked by reading, not run on hardware
//
#include <hip/hip_runtime.h>

// CDNA5 (gfx1250) wave32 WMMA causal self-attention, round 2.
// f32->f16 pre-convert pass, then three WMMA stages with pure-f16 inner loops:
//   1) qkv GEMM (32x64 tile/wave, 8 wmma : 12 b128-loads per K-step),
//      V written transposed [B,H,64,T] so the PV stage loads contiguously.
//   2) flash attention: S computed transposed (K as A, Q as B) so the f32
//      accumulator layout IS the f16 B-operand layout for P^T -> no lane
//      shuffles between softmax and the PV wmma.
//   3) output projection GEMM (f32 stores to d_out).

typedef __attribute__((ext_vector_type(16))) _Float16 v16h;
typedef __attribute__((ext_vector_type(8)))  _Float16 v8h;
typedef __attribute__((ext_vector_type(8)))  float    v8f;

#define WMMA_F16(A, B, Cacc) \
  __builtin_amdgcn_wmma_f32_16x16x32_f16(false, (A), false, (B), (short)0, (Cacc), false, false)

static constexpr int BB = 4;     // batch
static constexpr int TT = 2048;  // sequence
static constexpr int CC = 1024;  // channels
static constexpr int HH = 16;    // heads
static constexpr int HD = 64;    // head dim
static constexpr float SM_SCALE = 0.125f;  // 1/sqrt(64)
static constexpr float NEG_BIG = -1e30f;

// WMMA A/B operand gather from f16 row (lane = row/col, hs = lane>>4).
// ISA K-packing: idx 0..7 <-> K = 8*hs + idx ; idx 8..15 <-> K = 16 + 8*hs + idx-8.
// Both halves are contiguous 16B runs -> two b128 loads.
__device__ __forceinline__ v16h ld16_f16(const _Float16* __restrict__ p, int hs) {
  v16h r;
  v8h lo = *(const v8h*)(p + 8 * hs);
  v8h hi = *(const v8h*)(p + 16 + 8 * hs);
#pragma unroll
  for (int i = 0; i < 8; ++i) {
    r[i] = lo[i];
    r[8 + i] = hi[i];
  }
  return r;
}

// ---------------- Kernel 0: f32 -> f16 bulk convert (n multiple of 8)
__global__ __launch_bounds__(256) void cvt_f32_f16_kernel(
    const float* __restrict__ s, _Float16* __restrict__ d, int n) {
  int i = (blockIdx.x * 256 + threadIdx.x) * 8;
  if (i >= n) return;
  float4 a = *(const float4*)(s + i);
  float4 b = *(const float4*)(s + i + 4);
  v8h h;
  h[0] = (_Float16)a.x; h[1] = (_Float16)a.y; h[2] = (_Float16)a.z; h[3] = (_Float16)a.w;
  h[4] = (_Float16)b.x; h[5] = (_Float16)b.y; h[6] = (_Float16)b.z; h[7] = (_Float16)b.w;
  *(v8h*)(d + i) = h;
}

// ---------------- Kernel 1: qkv = x @ W_attn^T, scatter to Q/K [B,H,T,64] and
// V transposed [B,H,64,T], all f16. Wave tile 32(M) x 64(N).
__global__ __launch_bounds__(128) void qkv_gemm_kernel(
    const _Float16* __restrict__ Xh, const _Float16* __restrict__ Wh,
    _Float16* __restrict__ Q, _Float16* __restrict__ K, _Float16* __restrict__ Vt) {
  const int lane = threadIdx.x & 31;
  const int wave = threadIdx.x >> 5;
  const int l16 = lane & 15, hs = lane >> 4;
  const int m0 = blockIdx.x * 64 + (wave >> 1) * 32;   // rows (b*T+t)
  const int n0 = blockIdx.y * 128 + (wave & 1) * 64;   // cols (j), 64-aligned

  const _Float16* pa0 = Xh + (size_t)(m0 + l16) * CC;
  const _Float16* pa1 = pa0 + (size_t)16 * CC;
  const _Float16* pb0 = Wh + (size_t)(n0 + l16) * CC;
  const _Float16* pb1 = pb0 + (size_t)16 * CC;
  const _Float16* pb2 = pb0 + (size_t)32 * CC;
  const _Float16* pb3 = pb0 + (size_t)48 * CC;

  v8f a00{}, a01{}, a02{}, a03{}, a10{}, a11{}, a12{}, a13{};
  for (int c0 = 0; c0 < CC; c0 += 32) {
    v16h A0 = ld16_f16(pa0 + c0, hs);
    v16h A1 = ld16_f16(pa1 + c0, hs);
    v16h B0 = ld16_f16(pb0 + c0, hs);
    v16h B1 = ld16_f16(pb1 + c0, hs);
    v16h B2 = ld16_f16(pb2 + c0, hs);
    v16h B3 = ld16_f16(pb3 + c0, hs);
    a00 = WMMA_F16(A0, B0, a00);
    a01 = WMMA_F16(A0, B1, a01);
    a02 = WMMA_F16(A0, B2, a02);
    a03 = WMMA_F16(A0, B3, a03);
    a10 = WMMA_F16(A1, B0, a10);
    a11 = WMMA_F16(A1, B1, a11);
    a12 = WMMA_F16(A1, B2, a12);
    a13 = WMMA_F16(A1, B3, a13);
  }

  // The wave's 64-column span [n0, n0+64) maps to exactly one (head, kind):
  //   g = n0/64, h = g/3, kind = g%3  (qkv reshape [B,T,H,3*hd] semantics).
  const int g = n0 >> 6;
  const int h = g / 3;
  const int kind = g - 3 * h;
  _Float16* O = (kind == 0) ? Q : ((kind == 1) ? K : Vt);
  const bool vtrans = (kind == 2);

  // C/D layout: lane = N (column) -> d = (j%64); VGPR r -> M = r + 8*hs -> row i.
  auto store_tile = [&](const v8f& acc, int mt, int nt) {
    int d = (nt * 16 + l16);  // within-head channel 0..63
#pragma unroll
    for (int r = 0; r < 8; ++r) {
      int i = m0 + mt * 16 + 8 * hs + r;
      int b = i >> 11, t = i & 2047;
      size_t bh = (size_t)(b * HH + h);
      size_t idx = vtrans ? (bh * HD + d) * TT + t   // V^T: [B,H,64,T]
                          : (bh * TT + t) * HD + d;  // Q,K: [B,H,T,64]
      O[idx] = (_Float16)acc[r];
    }
  };
  store_tile(a00, 0, 0); store_tile(a01, 0, 1); store_tile(a02, 0, 2); store_tile(a03, 0, 3);
  store_tile(a10, 1, 0); store_tile(a11, 1, 1); store_tile(a12, 1, 2); store_tile(a13, 1, 3);
}

// ---------------- Kernel 2: flash attention, one wave per 16-query block.
__global__ __launch_bounds__(128) void flash_attn_kernel(
    const _Float16* __restrict__ Q, const _Float16* __restrict__ K,
    const _Float16* __restrict__ Vt, _Float16* __restrict__ Y) {
  const int lane = threadIdx.x & 31;
  const int wave = threadIdx.x >> 5;
  const int l16 = lane & 15, hs = lane >> 4;
  const int w = blockIdx.x * 4 + wave;
  const int qblk = w & 127;   // T/16 query blocks
  const int bh = w >> 7;      // b*H + h
  const int q0 = qblk * 16;

  const _Float16* Qp = Q + (size_t)bh * TT * HD;
  const _Float16* Kp = K + (size_t)bh * TT * HD;
  const _Float16* Vtp = Vt + (size_t)bh * HD * TT;  // [64, T]

  // Q block as B operand (lane = query column), hd=64 -> two K=32 chunks.
  const v16h qb0 = ld16_f16(Qp + (size_t)(q0 + l16) * HD, hs);
  const v16h qb1 = ld16_f16(Qp + (size_t)(q0 + l16) * HD + 32, hs);

  v8f ot0{}, ot1{}, ot2{}, ot3{};  // O^T: 4 d-groups x (16d x 16q)
  float mrow = NEG_BIG, lrow = 0.0f;
  const int qg = q0 + l16;  // this lane's global query index

  for (int j0 = 0; j0 <= q0 + 15; j0 += 32) {  // causal 32-key blocks
    v8f s0{}, s1{};
    {  // S^T = K(A) x Q(B): lanes = queries, VGPR r -> key = j0 + 8*hs + r (+16)
      v16h ka;
      ka = ld16_f16(Kp + (size_t)(j0 + l16) * HD, hs);           s0 = WMMA_F16(ka, qb0, s0);
      ka = ld16_f16(Kp + (size_t)(j0 + l16) * HD + 32, hs);      s0 = WMMA_F16(ka, qb1, s0);
      ka = ld16_f16(Kp + (size_t)(j0 + 16 + l16) * HD, hs);      s1 = WMMA_F16(ka, qb0, s1);
      ka = ld16_f16(Kp + (size_t)(j0 + 16 + l16) * HD + 32, hs); s1 = WMMA_F16(ka, qb1, s1);
    }

    // scale + causal mask + per-query (per-lane) max
    float p0[8], p1[8];
    float mloc = NEG_BIG;
#pragma unroll
    for (int r = 0; r < 8; ++r) {
      int k0 = j0 + 8 * hs + r;
      int k1 = k0 + 16;
      float x0 = (k0 <= qg) ? s0[r] * SM_SCALE : NEG_BIG;
      float x1 = (k1 <= qg) ? s1[r] * SM_SCALE : NEG_BIG;
      p0[r] = x0;
      p1[r] = x1;
      mloc = fmaxf(mloc, fmaxf(x0, x1));
    }
    mloc = fmaxf(mloc, __shfl_xor(mloc, 16, 32));  // combine key halves

    float mnew = fmaxf(mrow, mloc);
    float corr = __expf(mrow - mnew);
    float lloc = 0.0f;
    v16h pb;  // P^T packed directly into the f16 B-operand layout
#pragma unroll
    for (int r = 0; r < 8; ++r) {
      float e0 = __expf(p0[r] - mnew);
      float e1 = __expf(p1[r] - mnew);
      lloc += e0 + e1;
      pb[r] = (_Float16)e0;
      pb[8 + r] = (_Float16)e1;
    }
    lloc += __shfl_xor(lloc, 16, 32);
    lrow = lrow * corr + lloc;
    mrow = mnew;
#pragma unroll
    for (int r = 0; r < 8; ++r) {
      ot0[r] *= corr; ot1[r] *= corr; ot2[r] *= corr; ot3[r] *= corr;
    }

    // O^T += V^T(A) x P^T(B); V^T rows are contiguous in [64,T] layout.
    ot0 = WMMA_F16(ld16_f16(Vtp + (size_t)(0 * 16 + l16) * TT + j0, hs), pb, ot0);
    ot1 = WMMA_F16(ld16_f16(Vtp + (size_t)(1 * 16 + l16) * TT + j0, hs), pb, ot1);
    ot2 = WMMA_F16(ld16_f16(Vtp + (size_t)(2 * 16 + l16) * TT + j0, hs), pb, ot2);
    ot3 = WMMA_F16(ld16_f16(Vtp + (size_t)(3 * 16 + l16) * TT + j0, hs), pb, ot3);
  }

  const float inv = 1.0f / lrow;
  const int b = bh >> 4, h = bh & 15;
  // O^T layout: lane = query, VGPR r -> d = g*16 + 8*hs + r
  _Float16* Yp = Y + ((size_t)b * TT + (q0 + l16)) * CC + h * HD;
#pragma unroll
  for (int r = 0; r < 8; ++r) {
    Yp[0 * 16 + 8 * hs + r] = (_Float16)(ot0[r] * inv);
    Yp[1 * 16 + 8 * hs + r] = (_Float16)(ot1[r] * inv);
    Yp[2 * 16 + 8 * hs + r] = (_Float16)(ot2[r] * inv);
    Yp[3 * 16 + 8 * hs + r] = (_Float16)(ot3[r] * inv);
  }
}

// ---------------- Kernel 3: out = Y @ W_proj^T (f32 output), 32x64 tile/wave.
__global__ __launch_bounds__(128) void proj_gemm_kernel(
    const _Float16* __restrict__ Yh, const _Float16* __restrict__ Wh,
    float* __restrict__ Out) {
  const int lane = threadIdx.x & 31;
  const int wave = threadIdx.x >> 5;
  const int l16 = lane & 15, hs = lane >> 4;
  const int m0 = blockIdx.x * 64 + (wave >> 1) * 32;
  const int n0 = blockIdx.y * 128 + (wave & 1) * 64;

  const _Float16* pa0 = Yh + (size_t)(m0 + l16) * CC;
  const _Float16* pa1 = pa0 + (size_t)16 * CC;
  const _Float16* pb0 = Wh + (size_t)(n0 + l16) * CC;
  const _Float16* pb1 = pb0 + (size_t)16 * CC;
  const _Float16* pb2 = pb0 + (size_t)32 * CC;
  const _Float16* pb3 = pb0 + (size_t)48 * CC;

  v8f a00{}, a01{}, a02{}, a03{}, a10{}, a11{}, a12{}, a13{};
  for (int c0 = 0; c0 < CC; c0 += 32) {
    v16h A0 = ld16_f16(pa0 + c0, hs);
    v16h A1 = ld16_f16(pa1 + c0, hs);
    v16h B0 = ld16_f16(pb0 + c0, hs);
    v16h B1 = ld16_f16(pb1 + c0, hs);
    v16h B2 = ld16_f16(pb2 + c0, hs);
    v16h B3 = ld16_f16(pb3 + c0, hs);
    a00 = WMMA_F16(A0, B0, a00);
    a01 = WMMA_F16(A0, B1, a01);
    a02 = WMMA_F16(A0, B2, a02);
    a03 = WMMA_F16(A0, B3, a03);
    a10 = WMMA_F16(A1, B0, a10);
    a11 = WMMA_F16(A1, B1, a11);
    a12 = WMMA_F16(A1, B2, a12);
    a13 = WMMA_F16(A1, B3, a13);
  }

  auto store_tile = [&](const v8f& acc, int mt, int nt) {
    int j = n0 + nt * 16 + l16;
#pragma unroll
    for (int r = 0; r < 8; ++r) {
      int i = m0 + mt * 16 + 8 * hs + r;
      Out[(size_t)i * CC + j] = acc[r];
    }
  };
  store_tile(a00, 0, 0); store_tile(a01, 0, 1); store_tile(a02, 0, 2); store_tile(a03, 0, 3);
  store_tile(a10, 1, 0); store_tile(a11, 1, 1); store_tile(a12, 1, 2); store_tile(a13, 1, 3);
}

extern "C" void kernel_launch(void* const* d_in, const int* in_sizes, int n_in,
                              void* d_out, int out_size, void* d_ws, size_t ws_size,
                              hipStream_t stream) {
  (void)in_sizes; (void)n_in; (void)out_size; (void)ws_size;
  const float* X = (const float*)d_in[0];      // [4,2048,1024]
  const float* Wattn = (const float*)d_in[1];  // [3072,1024]
  const float* Wproj = (const float*)d_in[2];  // [1024,1024]
  float* Out = (float*)d_out;                  // [4,2048,1024]

  // Workspace (all f16):
  //   Xh 8.39M, Wah 3.15M, Wph 1.05M, Q/K/Vt 8.39M each, Y 8.39M  -> ~92 MB
  const size_t N_X = (size_t)BB * TT * CC;        // 8,388,608
  const size_t N_WA = (size_t)3 * CC * CC;        // 3,145,728
  const size_t N_WP = (size_t)CC * CC;            // 1,048,576
  const size_t N_QKV = (size_t)BB * HH * TT * HD; // 8,388,608

  _Float16* Xh = (_Float16*)d_ws;
  _Float16* Wah = Xh + N_X;
  _Float16* Wph = Wah + N_WA;
  _Float16* Qh = Wph + N_WP;
  _Float16* Kh = Qh + N_QKV;
  _Float16* Vth = Kh + N_QKV;
  _Float16* Yh = Vth + N_QKV;

  // 0) f32 -> f16 converts (bandwidth trivial vs 137 GFLOP of matmul)
  cvt_f32_f16_kernel<<<dim3((int)(N_X / (8 * 256))), dim3(256), 0, stream>>>(X, Xh, (int)N_X);
  cvt_f32_f16_kernel<<<dim3((int)(N_WA / (8 * 256))), dim3(256), 0, stream>>>(Wattn, Wah, (int)N_WA);
  cvt_f32_f16_kernel<<<dim3((int)(N_WP / (8 * 256))), dim3(256), 0, stream>>>(Wproj, Wph, (int)N_WP);

  // 1) QKV projection: 8192 x 3072, block covers 64(M) x 128(N)
  qkv_gemm_kernel<<<dim3((BB * TT) / 64, (3 * CC) / 128), dim3(128), 0, stream>>>(
      Xh, Wah, Qh, Kh, Vth);

  // 2) Flash attention: B*H*(T/16) waves, 4 waves/block
  flash_attn_kernel<<<dim3((BB * HH * (TT / 16)) / 4), dim3(128), 0, stream>>>(
      Qh, Kh, Vth, Yh);

  // 3) Output projection: 8192 x 1024
  proj_gemm_kernel<<<dim3((BB * TT) / 64, CC / 128), dim3(128), 0, stream>>>(
      Yh, Wproj ? Wph : Wph, Out);
}
